// MLA_8229157339841
// MI455X (gfx1250) — matrix operand order repaired and checked by reading, not compiled, not run on hardware
//
#include <hip/hip_runtime.h>
#include <hip/hip_bf16.h>

// ---------------------------------------------------------------------------
// MLA forward for MI455X (gfx1250, wave32, WMMA).
// All matmuls use v_wmma_f32_16x16x32_bf16 (bf16 operands, f32 accumulate).
// Tile staging uses GLOBAL_LOAD_ASYNC_TO_LDS_B128 (ASYNCcnt) with double
// buffering in the GEMM. Softmax / bias / RoPE in f32.
// ---------------------------------------------------------------------------

typedef __attribute__((ext_vector_type(16))) __bf16 v16bf;
typedef __attribute__((ext_vector_type(8)))  float  v8f;

#define D_MODEL 2048
#define N_HEADS 16
#define D_C     512
#define D_COMP  768
#define D_ROPE  64
#define D_HEAD  128
#define BATCH   2
#define SEQ     2048
#define MTOT    (BATCH * SEQ)   // 4096 rows

// f32 -> bf16 round-to-nearest-even
__device__ __forceinline__ __bf16 f2bf(float f) {
    union { float f; unsigned u; } in; in.f = f;
    unsigned u = in.u;
    u += 0x7FFFu + ((u >> 16) & 1u);
    union { unsigned short s; __bf16 b; } out;
    out.s = (unsigned short)(u >> 16);
    return out.b;
}

// ---- CDNA5 async global->LDS copy (16 bytes per lane, ASYNCcnt) -----------
// lds_addr: LDS byte offset (low 32 bits of generic shared address)
// voff:     per-lane byte offset added to sbase (GVS mode)
__device__ __forceinline__ void async_load_16B(unsigned lds_addr, unsigned voff,
                                               const void* sbase) {
    asm volatile("global_load_async_to_lds_b128 %0, %1, %2"
                 :: "v"(lds_addr), "v"(voff), "s"(sbase)
                 : "memory");
}
__device__ __forceinline__ void s_wait_async0() {
    asm volatile("s_wait_asynccnt 0x0" ::: "memory");
}
__device__ __forceinline__ void s_wait_async4() {
    asm volatile("s_wait_asynccnt 0x4" ::: "memory");
}
__device__ __forceinline__ unsigned lds_off(const void* p) {
    return (unsigned)(unsigned long long)p;   // LDS aperture: offset in addr[31:0]
}

// ---- WMMA fragment loaders (wave32; contiguous k = (lane>>4)*16 + j in both A and B) ----------
// A (16x32 bf16, row-major in LDS, leading dim ld):
//   lane m=lane&15, kb=(lane>>4)*16 ; a[j]=A[m][kb+j], a[8+j]=A[m][kb+8+j]
__device__ __forceinline__ v16bf load_a_frag(const __bf16* s, int ld) {
    const int lane = threadIdx.x & 31;
    const int m  = lane & 15;
    const int kb = (lane >> 4) * 16;
    v16bf a;
#pragma unroll
    for (int j = 0; j < 8; ++j) {
        a[j]     = s[m * ld + kb + j];
        a[j + 8] = s[m * ld + kb + 8 + j];
    }
    return a;
}
// B (32x16 bf16, stored row-major [k][n] with leading dim ld):
//   lane n=lane&15, kb=(lane>>4)*16 ; b[j]=B[kb+j][n]
__device__ __forceinline__ v16bf load_b_frag(const __bf16* s, int ld) {
    const int lane = threadIdx.x & 31;
    const int n  = lane & 15;
    const int kb = (lane >> 4) * 16;
    v16bf b;
#pragma unroll
    for (int j = 0; j < 16; ++j) b[j] = s[(kb + j) * ld + n];
    return b;
}
// B from transposed storage: s[n*ld + k] holds B[k][n]  (used for Q·K^T)
__device__ __forceinline__ v16bf load_bT_frag(const __bf16* s, int ld) {
    const int lane = threadIdx.x & 31;
    const int n  = lane & 15;
    const int kb = (lane >> 4) * 16;
    v16bf b;
#pragma unroll
    for (int j = 0; j < 16; ++j) b[j] = s[n * ld + kb + j];
    return b;
}

__device__ __forceinline__ v8f zero8() {
    v8f z = {0.f, 0.f, 0.f, 0.f, 0.f, 0.f, 0.f, 0.f};
    return z;
}

// ---------------------------------------------------------------------------
// f32 -> bf16 conversion kernel
// ---------------------------------------------------------------------------
__global__ void cvt_f32_bf16(const float* __restrict__ in, __bf16* __restrict__ out, int n) {
    int i = blockIdx.x * 256 + threadIdx.x;
    if (i < n) out[i] = f2bf(in[i]);
}

// ---------------------------------------------------------------------------
// Tiled WMMA GEMM: C[M,N] = A[M,K](bf16) @ B[K,N](bf16) + bias[N](f32)
// Block 256 threads (8 waves). Tile 128x128, BK=32, double-buffered LDS with
// async global->LDS DMA: next tile's 4 loads issued before waiting on the
// current tile (s_wait_asynccnt 4, in-order completion).
// Wave (wm in 0..3, wn in 0..1) owns a 32x64 region = 2x4 WMMA frags.
// ---------------------------------------------------------------------------
template <bool STORE_BF16>
__global__ __launch_bounds__(256)
void gemm_bf16_wmma(const __bf16* __restrict__ A, const __bf16* __restrict__ B,
                    const float* __restrict__ bias, void* __restrict__ Cout,
                    int M, int N, int K) {
    __shared__ __align__(16) __bf16 As[2][128 * 32];
    __shared__ __align__(16) __bf16 Bs[2][32 * 128];

    const int tid  = threadIdx.x;
    const int lane = tid & 31;
    const int wid  = tid >> 5;
    const int wm   = wid & 3;   // 4 waves along M (32 rows each)
    const int wn   = wid >> 2;  // 2 waves along N (64 cols each)
    const int bm   = blockIdx.y;
    const int bn   = blockIdx.x;

    v8f acc[2][4];
#pragma unroll
    for (int i = 0; i < 2; ++i)
#pragma unroll
        for (int j = 0; j < 4; ++j) acc[i][j] = zero8();

    // 16B chunks: A tile 128x32 = 512 chunks, B tile 32x128 = 512 chunks,
    // 2 of each per thread.
    const int ae0 = tid, ae1 = tid + 256;
    const int ar0 = ae0 >> 2, ac0 = (ae0 & 3) * 8;
    const int ar1 = ae1 >> 2, ac1 = (ae1 & 3) * 8;
    const int br0 = ae0 >> 4, bc0 = (ae0 & 15) * 8;
    const int br1 = ae1 >> 4, bc1 = (ae1 & 15) * 8;

    const unsigned ldsA = lds_off(&As[0][0]);
    const unsigned ldsB = lds_off(&Bs[0][0]);
    // per-lane global byte offsets, constant across the k loop
    const unsigned vA0 = (unsigned)((ar0 * K + ac0) * 2);
    const unsigned vA1 = (unsigned)((ar1 * K + ac1) * 2);
    const unsigned vB0 = (unsigned)((br0 * N + bn * 128 + bc0) * 2);
    const unsigned vB1 = (unsigned)((br1 * N + bn * 128 + bc1) * 2);
    const char* aBase = (const char*)(A + (size_t)bm * 128 * K);
    const char* bBase = (const char*)B;

    auto issue = [&](int k0, int buf) {
        const char* ab = aBase + (size_t)k0 * 2;
        const char* bb = bBase + (size_t)k0 * N * 2;
        const unsigned la = ldsA + (unsigned)buf * (128 * 32 * 2);
        const unsigned lb = ldsB + (unsigned)buf * (32 * 128 * 2);
        async_load_16B(la + (unsigned)(ar0 * 32 + ac0) * 2, vA0, ab);
        async_load_16B(la + (unsigned)(ar1 * 32 + ac1) * 2, vA1, ab);
        async_load_16B(lb + (unsigned)(br0 * 128 + bc0) * 2, vB0, bb);
        async_load_16B(lb + (unsigned)(br1 * 128 + bc1) * 2, vB1, bb);
    };

    const int nk = K / 32;
    issue(0, 0);
    for (int kt = 0; kt < nk; ++kt) {
        if (kt + 1 < nk) {
            issue((kt + 1) * 32, (kt + 1) & 1);
            s_wait_async4();   // oldest 4 (this tile) complete, next 4 in flight
        } else {
            s_wait_async0();
        }
        __syncthreads();       // tile kt visible to all waves

        const __bf16* as = &As[kt & 1][0];
        const __bf16* bs = &Bs[kt & 1][0];
        v16bf af[2], bf[4];
#pragma unroll
        for (int i = 0; i < 2; ++i)
            af[i] = load_a_frag(&as[(wm * 32 + i * 16) * 32], 32);
#pragma unroll
        for (int j = 0; j < 4; ++j)
            bf[j] = load_b_frag(&bs[wn * 64 + j * 16], 128);
#pragma unroll
        for (int i = 0; i < 2; ++i)
#pragma unroll
            for (int j = 0; j < 4; ++j)
                acc[i][j] = __builtin_amdgcn_wmma_f32_16x16x32_bf16(
                    false, af[i], false, bf[j], (short)0, acc[i][j], false, false);

        __syncthreads();       // all waves done reading buf kt&1
    }

    // store (C/D layout: row = (lane>>4)*8 + r, col = lane&15)
    const int r0 = (lane >> 4) * 8;
    const int cn = lane & 15;
#pragma unroll
    for (int i = 0; i < 2; ++i)
#pragma unroll
        for (int j = 0; j < 4; ++j)
#pragma unroll
            for (int r = 0; r < 8; ++r) {
                int row = bm * 128 + wm * 32 + i * 16 + r0 + r;
                int col = bn * 128 + wn * 64 + j * 16 + cn;
                float v = acc[i][j][r] + bias[col];
                if (STORE_BF16)
                    ((__bf16*)Cout)[(size_t)row * N + col] = f2bf(v);
                else
                    ((float*)Cout)[(size_t)row * N + col] = v;
            }
}

// ---------------------------------------------------------------------------
// RoPE: in f32 [B,S,H,64] -> out bf16, interleaved-pair rotation
// ---------------------------------------------------------------------------
__global__ void rope_kernel(const float* __restrict__ in, __bf16* __restrict__ out, int total) {
    int idx = blockIdx.x * 256 + threadIdx.x;
    if (idx >= total) return;
    int i = idx & 31;          // pair index 0..31
    int t = idx >> 5;          // flat (b, s, h)
    int s = (t >> 4) & (SEQ - 1);
    float ang = (float)s * __powf(10000.0f, -(float)i * (1.0f / 32.0f));
    float sn, cs;
    __sincosf(ang, &sn, &cs);
    const float* p = in + (size_t)t * 64;
    float x1 = p[2 * i], x2 = p[2 * i + 1];
    out[(size_t)t * 64 + 2 * i]     = f2bf(x1 * cs - x2 * sn);
    out[(size_t)t * 64 + 2 * i + 1] = f2bf(x1 * sn + x2 * cs);
}

// ---------------------------------------------------------------------------
// Flash attention, one (b, h, 64-row Q tile) per block. 128 threads = 4 waves;
// wave w owns Q rows [w*16, w*16+16). K tile = 64 keys x 192 dims staged via
// async global->LDS DMA; first 128 dims double as V. Online softmax in f32.
// ---------------------------------------------------------------------------
__global__ __launch_bounds__(128)
void mla_attn(const __bf16* __restrict__ qc, const __bf16* __restrict__ qr,
              const __bf16* __restrict__ kvup, const __bf16* __restrict__ kr,
              __bf16* __restrict__ out) {
    __shared__ __align__(16) __bf16 Qs[64 * 192];     // 24 KB
    __shared__ __align__(16) __bf16 Ks[64 * 192];     // 24 KB (cols 0..127 = V)
    __shared__ __align__(16) __bf16 Ps[4 * 16 * 64];  // 8 KB  (per-wave P)

    const int tid  = threadIdx.x;
    const int lane = tid & 31;
    const int wid  = tid >> 5;
    const int qt   = blockIdx.x;
    const int h    = blockIdx.y;
    const int b    = blockIdx.z;
    const size_t rowQ0 = (size_t)b * SEQ + qt * 64;

    // stage Q tile once: q_c (64x128) || q_r (64x64)
    for (int e = tid; e < 1024; e += 128) {
        int r = e >> 4, c = (e & 15) * 8;
        *(uint4*)&Qs[r * 192 + c] =
            *(const uint4*)&qc[(rowQ0 + r) * 2048 + h * 128 + c];
    }
    for (int e = tid; e < 512; e += 128) {
        int r = e >> 3, c = (e & 7) * 8;
        *(uint4*)&Qs[r * 192 + 128 + c] =
            *(const uint4*)&qr[(rowQ0 + r) * 1024 + h * 64 + c];
    }

    const unsigned ldsK = lds_off(&Ks[0]);
    const int qrow0 = wid * 16;
    float m_i[8], l_i[8];
    v8f o[8];
#pragma unroll
    for (int r = 0; r < 8; ++r) { m_i[r] = -1e30f; l_i[r] = 0.f; }
#pragma unroll
    for (int j = 0; j < 8; ++j) o[j] = zero8();

    const float scl = 0.08838834764831845f;  // 1/sqrt(128)

    for (int kt = 0; kt < SEQ / 64; ++kt) {
        __syncthreads();  // previous iteration done reading Ks
        const unsigned rowK0 = (unsigned)(b * SEQ + kt * 64);
        // kv_up part: 64x128 -> Ks cols 0..127 (8 chunks/thread)
        for (int e = tid; e < 1024; e += 128) {
            int r = e >> 4, c = (e & 15) * 8;
            async_load_16B(ldsK + (unsigned)(r * 192 + c) * 2,
                           ((rowK0 + r) * 2048u + h * 128u + c) * 2u, kvup);
        }
        // k_r part: 64x64 -> Ks cols 128..191 (4 chunks/thread)
        for (int e = tid; e < 512; e += 128) {
            int r = e >> 3, c = (e & 7) * 8;
            async_load_16B(ldsK + (unsigned)(r * 192 + 128 + c) * 2,
                           ((rowK0 + r) * 1024u + h * 64u + c) * 2u, kr);
        }
        s_wait_async0();
        __syncthreads();

        // S = Q . K^T  (16 q-rows x 64 keys, 6 k-steps over d=192)
        v8f sfr[4];
#pragma unroll
        for (int j = 0; j < 4; ++j) sfr[j] = zero8();
        for (int ks = 0; ks < 6; ++ks) {
            v16bf a = load_a_frag(&Qs[qrow0 * 192 + ks * 32], 192);
#pragma unroll
            for (int j = 0; j < 4; ++j) {
                v16bf bb = load_bT_frag(&Ks[(j * 16) * 192 + ks * 32], 192);
                sfr[j] = __builtin_amdgcn_wmma_f32_16x16x32_bf16(
                    false, a, false, bb, (short)0, sfr[j], false, false);
            }
        }
#pragma unroll
        for (int j = 0; j < 4; ++j)
#pragma unroll
            for (int r = 0; r < 8; ++r) sfr[j][r] *= scl;

        // online softmax (rows live in 16-lane halves: shfl_xor 1,2,4,8)
        float rmax[8];
#pragma unroll
        for (int r = 0; r < 8; ++r) {
            float v = sfr[0][r];
#pragma unroll
            for (int j = 1; j < 4; ++j) v = fmaxf(v, sfr[j][r]);
            rmax[r] = v;
        }
#pragma unroll
        for (int r = 0; r < 8; ++r) {
#pragma unroll
            for (int msk = 1; msk < 16; msk <<= 1)
                rmax[r] = fmaxf(rmax[r], __shfl_xor(rmax[r], msk, 32));
        }
        float fct[8], rsum[8];
#pragma unroll
        for (int r = 0; r < 8; ++r) {
            float mn = fmaxf(m_i[r], rmax[r]);
            fct[r] = __expf(m_i[r] - mn);
            m_i[r] = mn;
            rsum[r] = 0.f;
        }
#pragma unroll
        for (int j = 0; j < 4; ++j)
#pragma unroll
            for (int r = 0; r < 8; ++r) {
                float p = __expf(sfr[j][r] - m_i[r]);
                sfr[j][r] = p;
                rsum[r] += p;
            }
#pragma unroll
        for (int r = 0; r < 8; ++r) {
#pragma unroll
            for (int msk = 1; msk < 16; msk <<= 1)
                rsum[r] += __shfl_xor(rsum[r], msk, 32);
            l_i[r] = l_i[r] * fct[r] + rsum[r];
        }
#pragma unroll
        for (int j = 0; j < 8; ++j)
#pragma unroll
            for (int r = 0; r < 8; ++r) o[j][r] *= fct[r];

        // repack P (C-layout -> A-layout) through per-wave LDS
        __bf16* Pw = &Ps[wid * 16 * 64];
        const int pr0 = (lane >> 4) * 8, pc = lane & 15;
#pragma unroll
        for (int j = 0; j < 4; ++j)
#pragma unroll
            for (int r = 0; r < 8; ++r)
                Pw[(pr0 + r) * 64 + j * 16 + pc] = f2bf(sfr[j][r]);
        __syncthreads();

        // O += P . V  (V = Ks cols 0..127), 2 k-steps over 64 keys
#pragma unroll
        for (int ks2 = 0; ks2 < 2; ++ks2) {
            v16bf a = load_a_frag(&Pw[ks2 * 32], 64);
#pragma unroll
            for (int j = 0; j < 8; ++j) {
                v16bf bb = load_b_frag(&Ks[(ks2 * 32) * 192 + j * 16], 192);
                o[j] = __builtin_amdgcn_wmma_f32_16x16x32_bf16(
                    false, a, false, bb, (short)0, o[j], false, false);
            }
        }
    }

    // epilogue: normalize and store bf16 [row, h*128 + d]
    const int r0 = (lane >> 4) * 8, cn = lane & 15;
#pragma unroll
    for (int r = 0; r < 8; ++r) {
        float inv = 1.0f / l_i[r];
#pragma unroll
        for (int j = 0; j < 8; ++j) {
            size_t row = rowQ0 + qrow0 + r0 + r;
            out[row * 2048 + h * 128 + j * 16 + cn] = f2bf(o[j][r] * inv);
        }
    }
}

// ---------------------------------------------------------------------------
extern "C" void kernel_launch(void* const* d_in, const int* in_sizes, int n_in,
                              void* d_out, int out_size, void* d_ws, size_t ws_size,
                              hipStream_t stream) {
    (void)in_sizes; (void)n_in; (void)out_size; (void)ws_size;
    const float* x   = (const float*)d_in[0];
    const float* Wd  = (const float*)d_in[1];
    const float* bd  = (const float*)d_in[2];
    const float* Wu  = (const float*)d_in[3];
    const float* bu  = (const float*)d_in[4];
    const float* Wqd = (const float*)d_in[5];
    const float* bqd = (const float*)d_in[6];
    const float* Wqu = (const float*)d_in[7];
    const float* bqu = (const float*)d_in[8];
    const float* Wqr = (const float*)d_in[9];
    const float* bqr = (const float*)d_in[10];
    const float* Wkr = (const float*)d_in[11];
    const float* bkr = (const float*)d_in[12];
    const float* Wo  = (const float*)d_in[13];
    const float* bo  = (const float*)d_in[14];

    char* ws = (char*)d_ws;
    size_t off = 0;
    auto take = [&](size_t bytes) -> char* {
        char* p = ws + off;
        off += (bytes + 255) & ~(size_t)255;
        return p;
    };

    __bf16* xb    = (__bf16*)take((size_t)MTOT * D_MODEL * 2);
    __bf16* Wdb   = (__bf16*)take((size_t)D_MODEL * D_C * 2);
    __bf16* Wub   = (__bf16*)take((size_t)D_C * D_MODEL * 2);
    __bf16* Wqdb  = (__bf16*)take((size_t)D_MODEL * D_COMP * 2);
    __bf16* Wqub  = (__bf16*)take((size_t)D_COMP * D_MODEL * 2);
    __bf16* Wqrb  = (__bf16*)take((size_t)D_COMP * N_HEADS * D_ROPE * 2);
    __bf16* Wkrb  = (__bf16*)take((size_t)D_MODEL * N_HEADS * D_ROPE * 2);
    __bf16* Wob   = (__bf16*)take((size_t)D_MODEL * D_MODEL * 2);
    __bf16* kvcb  = (__bf16*)take((size_t)MTOT * D_C * 2);
    __bf16* kvupb = (__bf16*)take((size_t)MTOT * D_MODEL * 2);
    __bf16* qcmpb = (__bf16*)take((size_t)MTOT * D_COMP * 2);
    __bf16* qcb   = (__bf16*)take((size_t)MTOT * D_MODEL * 2);
    float*  qrl   = (float*)take((size_t)MTOT * 1024 * 4);
    float*  krl   = (float*)take((size_t)MTOT * 1024 * 4);
    __bf16* qrb   = (__bf16*)take((size_t)MTOT * 1024 * 2);
    __bf16* krb   = (__bf16*)take((size_t)MTOT * 1024 * 2);
    __bf16* attnb = (__bf16*)take((size_t)MTOT * D_MODEL * 2);

    auto cvt = [&](const float* src, __bf16* dst, size_t n) {
        cvt_f32_bf16<<<(unsigned)((n + 255) / 256), 256, 0, stream>>>(src, dst, (int)n);
    };
    cvt(x,   xb,   (size_t)MTOT * D_MODEL);
    cvt(Wd,  Wdb,  (size_t)D_MODEL * D_C);
    cvt(Wu,  Wub,  (size_t)D_C * D_MODEL);
    cvt(Wqd, Wqdb, (size_t)D_MODEL * D_COMP);
    cvt(Wqu, Wqub, (size_t)D_COMP * D_MODEL);
    cvt(Wqr, Wqrb, (size_t)D_COMP * 1024);
    cvt(Wkr, Wkrb, (size_t)D_MODEL * 1024);
    cvt(Wo,  Wob,  (size_t)D_MODEL * D_MODEL);

    // projections (all dims are multiples of 128/32)
    gemm_bf16_wmma<true><<<dim3(D_C / 128, MTOT / 128), 256, 0, stream>>>(
        xb, Wdb, bd, kvcb, MTOT, D_C, D_MODEL);
    gemm_bf16_wmma<true><<<dim3(D_MODEL / 128, MTOT / 128), 256, 0, stream>>>(
        kvcb, Wub, bu, kvupb, MTOT, D_MODEL, D_C);
    gemm_bf16_wmma<true><<<dim3(D_COMP / 128, MTOT / 128), 256, 0, stream>>>(
        xb, Wqdb, bqd, qcmpb, MTOT, D_COMP, D_MODEL);
    gemm_bf16_wmma<true><<<dim3(D_MODEL / 128, MTOT / 128), 256, 0, stream>>>(
        qcmpb, Wqub, bqu, qcb, MTOT, D_MODEL, D_COMP);
    gemm_bf16_wmma<false><<<dim3(1024 / 128, MTOT / 128), 256, 0, stream>>>(
        qcmpb, Wqrb, bqr, (void*)qrl, MTOT, 1024, D_COMP);
    gemm_bf16_wmma<false><<<dim3(1024 / 128, MTOT / 128), 256, 0, stream>>>(
        xb, Wkrb, bkr, (void*)krl, MTOT, 1024, D_MODEL);

    // RoPE on q_r / k_r
    const int rope_total = MTOT * N_HEADS * (D_ROPE / 2);  // 2M pairs
    rope_kernel<<<(rope_total + 255) / 256, 256, 0, stream>>>(qrl, qrb, rope_total);
    rope_kernel<<<(rope_total + 255) / 256, 256, 0, stream>>>(krl, krb, rope_total);

    // attention
    mla_attn<<<dim3(SEQ / 64, N_HEADS, BATCH), 128, 0, stream>>>(
        qcb, qrb, kvupb, krb, attnb);

    // output projection -> f32 d_out
    gemm_bf16_wmma<false><<<dim3(D_MODEL / 128, MTOT / 128), 256, 0, stream>>>(
        attnb, Wob, bo, d_out, MTOT, D_MODEL, D_MODEL);
}
